// TransformerBlock_46523085750728
// MI455X (gfx1250) — matrix operand order repaired and checked by reading, not compiled, not run on hardware
//
#include <hip/hip_runtime.h>
#include <hip/hip_bf16.h>
#include <math.h>

typedef __attribute__((ext_vector_type(16))) _Float16 v16h;
typedef __attribute__((ext_vector_type(8)))  _Float16 v8h;
typedef __attribute__((ext_vector_type(8)))  float    v8f;

#define HWSZ 36864      // 192*192
#define WDIM 192
#define NPIX (256.0f * 36864.0f)

// Build a v16h fragment from two contiguous 16-byte LDS runs (ds_load_b128 x2).
__device__ __forceinline__ v16h frag2(const _Float16* p0, const _Float16* p1) {
    v8h lo = *(const v8h*)p0;
    v8h hi = *(const v8h*)p1;
    return __builtin_shufflevector(lo, hi, 0, 1, 2, 3, 4, 5, 6, 7,
                                   8, 9, 10, 11, 12, 13, 14, 15);
}

// ---------------------------------------------------------------------------
// init: zero the stats slots in workspace (graph replays must not accumulate)
// ---------------------------------------------------------------------------
__global__ void init_ws_kernel(float* ws) {
    if (threadIdx.x < 16) ws[threadIdx.x] = 0.0f;
}

// ---------------------------------------------------------------------------
// weight prep: fp32 -> f16 casts, SAME layout as the reference weights:
//   w1h[o*256 + k]  (o=hidden 0..1023, k=in-ch 0..255)   == B-operand row-major
//   w2h[o*1024 + k] (o=out-ch 0..255,  k=hidden 0..1023) == B-operand row-major
// ---------------------------------------------------------------------------
__global__ void prep_weights_kernel(const float* __restrict__ w1,
                                    const float* __restrict__ w2,
                                    _Float16* __restrict__ w1h,
                                    _Float16* __restrict__ w2h) {
    int idx = blockIdx.x * blockDim.x + threadIdx.x;   // 262144 threads
    w1h[idx] = (_Float16)w1[idx];
    w2h[idx] = (_Float16)w2[idx];
}

// ---------------------------------------------------------------------------
// relative-position bias table: relb[h][n][m], h<8, n,m<64
// ---------------------------------------------------------------------------
__global__ void relbias_kernel(const float* __restrict__ w1, const float* __restrict__ b1,
                               const float* __restrict__ w2, const float* __restrict__ b2,
                               float* __restrict__ relb) {
    int idx = blockIdx.x * blockDim.x + threadIdx.x;   // 0..4095
    int n = idx >> 6, m = idx & 63;
    float dy = (float)(n >> 3) - (float)(m >> 3);
    float dx = (float)(n & 7)  - (float)(m & 7);
    float r0 = (dy > 0.f ? 1.f : (dy < 0.f ? -1.f : 0.f)) * log1pf(fabsf(dy));
    float r1 = (dx > 0.f ? 1.f : (dx < 0.f ? -1.f : 0.f)) * log1pf(fabsf(dx));
    float acc[8];
#pragma unroll
    for (int h = 0; h < 8; ++h) acc[h] = 0.f;
    for (int i = 0; i < 256; ++i) {
        float hv = r0 * w1[i] + r1 * w1[256 + i] + b1[i];
        hv = fmaxf(hv, 0.f);
#pragma unroll
        for (int h = 0; h < 8; ++h) acc[h] += hv * w2[i * 8 + h];
    }
#pragma unroll
    for (int h = 0; h < 8; ++h) relb[h * 4096 + idx] = acc[h] + b2[h];
}

// ---------------------------------------------------------------------------
// per-batch sum / sumsq reduction (RLN statistics): grid (256, B), block 256
// ---------------------------------------------------------------------------
__global__ void __launch_bounds__(256) stats_kernel(const float* __restrict__ src,
                                                    float* __restrict__ stats) {
    const int b = blockIdx.y;
    const long long base = (long long)b * (256LL * HWSZ);
    const long long tot = 256LL * HWSZ;
    float s = 0.f, q = 0.f;
    for (long long i = (long long)blockIdx.x * blockDim.x + threadIdx.x; i < tot;
         i += (long long)gridDim.x * blockDim.x) {
        float v = src[base + i];
        s += v; q += v * v;
    }
    __shared__ float rs[256], rq[256];
    rs[threadIdx.x] = s; rq[threadIdx.x] = q;
    __syncthreads();
    for (int off = 128; off > 0; off >>= 1) {
        if (threadIdx.x < off) {
            rs[threadIdx.x] += rs[threadIdx.x + off];
            rq[threadIdx.x] += rq[threadIdx.x + off];
        }
        __syncthreads();
    }
    if (threadIdx.x == 0) {
        atomicAdd(&stats[b * 2 + 0], rs[0]);
        atomicAdd(&stats[b * 2 + 1], rq[0]);
    }
}

// ---------------------------------------------------------------------------
// window attention: 1 block per 8x8 window, 8 waves cooperate per head.
// Writes stage-1 result y = x + attn*rescale + rebias  (y == d_out).
// ---------------------------------------------------------------------------
__global__ void __launch_bounds__(256) attn_kernel(
    const float* __restrict__ x, const float* __restrict__ stats,
    const float* __restrict__ relb,
    const float* __restrict__ n1w, const float* __restrict__ n1b,
    const float* __restrict__ m1w, const float* __restrict__ m1b,
    const float* __restrict__ m2w, const float* __restrict__ m2b,
    float* __restrict__ y) {
    __shared__ __align__(16) _Float16 Qlds[64 * 40];  // [token][dim],  pitch 40
    __shared__ __align__(16) _Float16 Qt[32 * 72];    // [dim][token],  pitch 72
    __shared__ float    Slds[64 * 65];                // scores f32,    pitch 65
    __shared__ __align__(16) _Float16 Plds[64 * 72];  // probs f16,     pitch 72

    const int blk = blockIdx.x;
    const int b   = blk / 576;
    const int rem = blk % 576;
    const int h0  = (rem / 24) * 8;
    const int w0  = (rem % 24) * 8;
    const int tid = threadIdx.x;
    const int wid = tid >> 5, lane = tid & 31;
    const int lh = lane & 15, hi = lane >> 4;

    const float mean = stats[b * 2 + 0] / NPIX;
    float var = stats[b * 2 + 1] / NPIX - mean * mean;
    var = var < 0.f ? 0.f : var;
    const float stdv = sqrtf(var) + 1e-5f;
    const float inv  = 1.0f / stdv;
    const float scale = 0.17677669529663687f;   // 32^-0.5
    const long long xbase = (long long)b * 256 * HWSZ;

    for (int head = 0; head < 8; ++head) {
        // ---- stage normalized Q (f16), both orientations ----
        for (int idx = tid; idx < 2048; idx += 256) {
            int d = idx >> 6, t = idx & 63;
            int c = head * 32 + d;
            long long g = xbase + (long long)c * HWSZ +
                          (long long)(h0 + (t >> 3)) * WDIM + (w0 + (t & 7));
            float v = (x[g] - mean) * inv * n1w[c] + n1b[c];
            _Float16 vh = (_Float16)v;
            Qlds[t * 40 + d] = vh;
            Qt[d * 72 + t]   = vh;
        }
        __syncthreads();

        // ---- S = Q Q^T  (16 tiles, 2 per wave) ----
        for (int t2 = 0; t2 < 2; ++t2) {
            int tile = wid * 2 + t2;
            int mb = (tile >> 2) * 16, nb = (tile & 3) * 16;
            const _Float16* ar = &Qlds[(mb + lh) * 40 + hi * 8];
            v16h a  = frag2(ar, ar + 16);
            const _Float16* br = &Qlds[(nb + lh) * 40 + hi * 8];
            v16h bb = frag2(br, br + 16);
            v8f c = {};
            c = __builtin_amdgcn_wmma_f32_16x16x32_f16(false, a, false, bb,
                                                       (short)0, c, false, false);
#pragma unroll
            for (int r = 0; r < 8; ++r)
                Slds[(mb + r + hi * 8) * 65 + nb + lh] = c[r];
        }
        __syncthreads();

        // ---- softmax rows (one row per thread, tid<64) ----
        if (tid < 64) {
            const float* bias = relb + head * 4096 + tid * 64;
            float mx = -3.0e38f;
            for (int c = 0; c < 64; ++c) {
                float v = Slds[tid * 65 + c] * scale + bias[c];
                mx = fmaxf(mx, v);
            }
            float sum = 0.f;
            for (int c = 0; c < 64; ++c) {
                float v = Slds[tid * 65 + c] * scale + bias[c];
                float e = expf(v - mx);
                Slds[tid * 65 + c] = e;
                sum += e;
            }
            float isum = 1.0f / sum;
            for (int c = 0; c < 64; ++c)
                Plds[tid * 72 + c] = (_Float16)(Slds[tid * 65 + c] * isum);
        }
        __syncthreads();

        // ---- O = P Q  (8 tiles, 1 per wave, K=64 -> 2 wmma) + epilogue ----
        {
            int mb = (wid & 3) * 16, nb = (wid >> 2) * 16;
            v8f o = {};
#pragma unroll
            for (int s = 0; s < 2; ++s) {
                const _Float16* ar = &Plds[(mb + lh) * 72 + s * 32 + hi * 8];
                v16h a  = frag2(ar, ar + 16);
                const _Float16* br = &Qt[(nb + lh) * 72 + s * 32 + hi * 8];
                v16h bb = frag2(br, br + 16);
                o = __builtin_amdgcn_wmma_f32_16x16x32_f16(false, a, false, bb,
                                                           (short)0, o, false, false);
            }
            int c = head * 32 + nb + lh;
            float rsc = stdv * m1w[c] + m1b[c];
            float rbs = mean * m2w[c] + m2b[c];
#pragma unroll
            for (int r = 0; r < 8; ++r) {
                int t = mb + r + hi * 8;
                long long g = xbase + (long long)c * HWSZ +
                              (long long)(h0 + (t >> 3)) * WDIM + (w0 + (t & 7));
                y[g] = x[g] + o[r] * rsc + rbs;
            }
        }
        __syncthreads();
    }
}

// ---------------------------------------------------------------------------
// MLP: 1 block per 64 pixels. X tile pixel-major f16 in LDS; hidden dim in
// 32-wide chunks; W1/W2 chunks time-share one LDS buffer. In-place update of y.
// ---------------------------------------------------------------------------
__global__ void __launch_bounds__(256) mlp_kernel(
    float* __restrict__ y, const float* __restrict__ stats,
    const _Float16* __restrict__ w1h, const float* __restrict__ b1,
    const _Float16* __restrict__ w2h, const float* __restrict__ b2,
    const float* __restrict__ n2w, const float* __restrict__ n2b,
    const float* __restrict__ m1w, const float* __restrict__ m1b,
    const float* __restrict__ m2w, const float* __restrict__ m2b) {
    __shared__ __align__(16) _Float16 Xp[64 * 264];   // [pix][ch],   pitch 264
    __shared__ __align__(16) _Float16 wbuf[10240];    // W1c [32][264] / W2c [256][40]
    __shared__ __align__(16) _Float16 Hlds[64 * 40];  // [pix][hid32], pitch 40

    const int blk = blockIdx.x;          // 1152 blocks
    const int b = blk / 576;
    const int p0 = (blk % 576) * 64;
    const int tid = threadIdx.x;
    const int wid = tid >> 5, lane = tid & 31;
    const int lh = lane & 15, hi = lane >> 4;

    const float mean = stats[b * 2 + 0] / NPIX;
    float var = stats[b * 2 + 1] / NPIX - mean * mean;
    var = var < 0.f ? 0.f : var;
    const float stdv = sqrtf(var) + 1e-5f;
    const float inv  = 1.0f / stdv;
    const long long base = (long long)b * 256 * HWSZ + p0;

    // ---- stage normalized input tile (f16, pixel-major) ----
    for (int it = 0; it < 64; ++it) {
        int c = it * 4 + (tid >> 6);
        int p = tid & 63;
        float v = (y[base + (long long)c * HWSZ + p] - mean) * inv * n2w[c] + n2b[c];
        Xp[p * 264 + c] = (_Float16)v;
    }
    __syncthreads();

    v8f acc[8];
    {
        v8f z = {};
#pragma unroll
        for (int i = 0; i < 8; ++i) acc[i] = z;
    }

    for (int cc = 0; cc < 32; ++cc) {
        const int hb = cc * 32;
        // ---- W1 chunk, hidden-major: wbuf[j*264 + k] = w1h[(hb+j)*256 + k] ----
        for (int idx = tid; idx < 8192; idx += 256) {
            int j = idx >> 8, k = idx & 255;
            wbuf[j * 264 + k] = w1h[(hb + j) * 256 + k];
        }
        __syncthreads();

        // ---- H = relu(Xn @ W1c + b1): 8 tiles, 1/wave, 8 k-steps ----
        {
            int m = wid & 3, nj = wid >> 2;
            v8f hacc = {};
#pragma unroll
            for (int ks = 0; ks < 8; ++ks) {
                const _Float16* ar = &Xp[(m * 16 + lh) * 264 + ks * 32 + hi * 8];
                v16h a  = frag2(ar, ar + 16);
                const _Float16* br = &wbuf[(nj * 16 + lh) * 264 + ks * 32 + hi * 8];
                v16h bb = frag2(br, br + 16);
                hacc = __builtin_amdgcn_wmma_f32_16x16x32_f16(false, a, false, bb,
                                                              (short)0, hacc, false, false);
            }
            float b1v = b1[hb + nj * 16 + lh];
#pragma unroll
            for (int r = 0; r < 8; ++r) {
                float hv = hacc[r] + b1v;
                hv = hv < 0.f ? 0.f : hv;
                Hlds[(m * 16 + r + hi * 8) * 40 + nj * 16 + lh] = (_Float16)hv;
            }
        }
        __syncthreads();

        // ---- W2 chunk, outch-major: wbuf[n*40 + k] = w2h[n*1024 + hb + k] ----
        for (int idx = tid; idx < 8192; idx += 256) {
            int n = idx >> 5, k = idx & 31;
            wbuf[n * 40 + k] = w2h[n * 1024 + hb + k];
        }
        if (cc + 1 < 32) {
            // prefetch next chunk's W1 rows (lowers to global_prefetch_b8)
            __builtin_prefetch(&w1h[(hb + 32) * 256] + tid * 32, 0, 1);
        }
        __syncthreads();

        // ---- acc += H @ W2c: wave owns out cols [wid*32, wid*32+32) ----
#pragma unroll
        for (int m = 0; m < 4; ++m) {
#pragma unroll
            for (int nn = 0; nn < 2; ++nn) {
                const _Float16* ar = &Hlds[(m * 16 + lh) * 40 + hi * 8];
                v16h a  = frag2(ar, ar + 16);
                const _Float16* br = &wbuf[(wid * 32 + nn * 16 + lh) * 40 + hi * 8];
                v16h bb = frag2(br, br + 16);
                acc[m * 2 + nn] = __builtin_amdgcn_wmma_f32_16x16x32_f16(
                    false, a, false, bb, (short)0, acc[m * 2 + nn], false, false);
            }
        }
        __syncthreads();
    }

    // ---- epilogue: y = identity + (mlp + b2)*rescale + rebias (in place) ----
#pragma unroll
    for (int nn = 0; nn < 2; ++nn) {
        int c = wid * 32 + nn * 16 + lh;
        float rsc = stdv * m1w[c] + m1b[c];
        float rbs = mean * m2w[c] + m2b[c];
        float b2v = b2[c];
#pragma unroll
        for (int m = 0; m < 4; ++m) {
#pragma unroll
            for (int r = 0; r < 8; ++r) {
                int p = m * 16 + r + hi * 8;
                long long g = base + (long long)c * HWSZ + p;
                float idv = y[g];
                y[g] = idv + (acc[m * 2 + nn][r] + b2v) * rsc + rbs;
            }
        }
    }
}

// ---------------------------------------------------------------------------
// launcher
// Workspace layout (bytes):  [0,64)    stats1 / stats2 (float idx 0..3, 8..11)
//                            [256, +131072)   rel-bias table f32 [8][64][64]
//                            [131584, +524288) w1h f16 [1024][256]
//                            [655872, +524288) w2h f16 [256][1024]
// Total ~1.13 MB of d_ws.  d_out doubles as the 75.5 MB stage-1 intermediate.
// ---------------------------------------------------------------------------
extern "C" void kernel_launch(void* const* d_in, const int* in_sizes, int n_in,
                              void* d_out, int out_size, void* d_ws, size_t ws_size,
                              hipStream_t stream) {
    (void)in_sizes; (void)n_in; (void)out_size; (void)ws_size;
    const float* x      = (const float*)d_in[0];
    const float* n1_w   = (const float*)d_in[1];
    const float* n1_b   = (const float*)d_in[2];
    const float* n1_m1w = (const float*)d_in[3];
    const float* n1_m1b = (const float*)d_in[4];
    const float* n1_m2w = (const float*)d_in[5];
    const float* n1_m2b = (const float*)d_in[6];
    const float* a_w1   = (const float*)d_in[7];
    const float* a_b1   = (const float*)d_in[8];
    const float* a_w2   = (const float*)d_in[9];
    const float* a_b2   = (const float*)d_in[10];
    const float* n2_w   = (const float*)d_in[11];
    const float* n2_b   = (const float*)d_in[12];
    const float* n2_m1w = (const float*)d_in[13];
    const float* n2_m1b = (const float*)d_in[14];
    const float* n2_m2w = (const float*)d_in[15];
    const float* n2_m2b = (const float*)d_in[16];
    const float* mlp_w1 = (const float*)d_in[17];
    const float* mlp_b1 = (const float*)d_in[18];
    const float* mlp_w2 = (const float*)d_in[19];
    const float* mlp_b2 = (const float*)d_in[20];

    float* ws_f   = (float*)d_ws;
    float* stats1 = ws_f;                         // floats [0..3]
    float* stats2 = ws_f + 8;                     // floats [8..11]
    float* relb   = ws_f + 64;                    // 32768 floats
    _Float16* w1h = (_Float16*)((char*)d_ws + 131584);
    _Float16* w2h = (_Float16*)((char*)d_ws + 131584 + 524288);
    float* y = (float*)d_out;

    init_ws_kernel<<<1, 32, 0, stream>>>(ws_f);
    prep_weights_kernel<<<1024, 256, 0, stream>>>(mlp_w1, mlp_w2, w1h, w2h);
    relbias_kernel<<<16, 256, 0, stream>>>(a_w1, a_b1, a_w2, a_b2, relb);

    stats_kernel<<<dim3(256, 2), 256, 0, stream>>>(x, stats1);
    attn_kernel<<<1152, 256, 0, stream>>>(x, stats1, relb, n1_w, n1_b,
                                          n1_m1w, n1_m1b, n1_m2w, n1_m2b, y);
    stats_kernel<<<dim3(256, 2), 256, 0, stream>>>(y, stats2);
    mlp_kernel<<<1152, 256, 0, stream>>>(y, stats2, w1h, mlp_b1, w2h, mlp_b2,
                                         n2_w, n2_b, n2_m1w, n2_m1b, n2_m2w, n2_m2b);
}
